// AutoCorrelation_Attention_3032246911663
// MI455X (gfx1250) — matrix operand order repaired and checked by reading, not compiled, not run on hardware
//
#include <hip/hip_runtime.h>

// ---------------------------------------------------------------------------
// AutoCorrelation attention for MI455X (gfx1250, wave32, WMMA).
//
//   1) q/k/v projections: bf16 WMMA GEMM (M=32768, N=K=1024), outputs written
//      fp32 in [B,H,hd,L] layout (contiguous per-series for the FFT stage).
//   2) per-series fused kernel: one complex FFT of (q + i*k) in LDS,
//      Hermitian split, P = Q*conj(K), inverse FFT -> corr, top-7 delays,
//      softmax, circular gather of v -> agg written as *bf16* [B,H,hd,L].
//   3) output projection: bf16 WMMA GEMM reading bf16 agg with [B,H,hd,L]
//      strides, writing [B,L,D] fp32 to d_out.
//
// GEMM geometry: block tile 128(M) x 256(N) x 32(K), 8 waves in 2x4 grid,
// 64x64 per wave = 16 WMMA vs 16 ds_load_b128 per K-step (1.0 loads/WMMA).
//
// Workspace (>= 3 * 16*1024*2048 * 4B = 384 MB): qbuf | kbuf | vbuf (fp32).
// agg (bf16) overwrites qbuf (per-block read-before-write alias only).
// ---------------------------------------------------------------------------

typedef __attribute__((ext_vector_type(16))) __bf16 v16bf;
typedef __attribute__((ext_vector_type(8)))  float  v8f;

union BFrag { v16bf bf; unsigned short u[16]; uint4 q[2]; };
union U4U   { uint4 v; unsigned short us[8]; };

__device__ __forceinline__ unsigned short f2bf(float f) {
    unsigned int u = __builtin_bit_cast(unsigned int, f);
    unsigned int r = u + 0x7FFFu + ((u >> 16) & 1u);   // round-to-nearest-even
    return (unsigned short)(r >> 16);
}
__device__ __forceinline__ unsigned int pack2bf(float lo, float hi) {
    return (unsigned int)f2bf(lo) | ((unsigned int)f2bf(hi) << 16);
}

// ------------------------------ WMMA GEMM ----------------------------------
// C[m,n] = sum_k A[m,k]*W[k,n] + bias[n],  M = 32768, N = K = 1024.
// AMODE 0: A fp32 row-major [M,1024].
// AMODE 1: A bf16, A[m,k] = buf[(b*1024 + k)*2048 + l], b=m>>11, l=m&2047.
// OMODE 0: Out[m*1024 + n].   OMODE 1: Out[(b*1024 + n)*2048 + l].
#define BM 128
#define BN 256
#define BKK 32
#define LDR 40   // padded LDS row stride in halfwords (80B, 16B-multiple)

template<int AMODE, int OMODE>
__global__ __launch_bounds__(256)
void wmma_gemm_bias(const void* Av, const float* W, const float* bias,
                    float* Out)
{
    __shared__ alignas(16) unsigned short As [BM * LDR];  // [m][k]
    __shared__ alignas(16) unsigned short BsT[BN * LDR];  // [n][k] (transposed)

    const int tid  = threadIdx.x;
    const int lane = tid & 31;
    const int wave = tid >> 5;
    const int wm   = wave >> 2;          // 0..1  (M, 64 rows each)
    const int wn   = wave & 3;           // 0..3  (N, 64 cols each)
    const int m0   = blockIdx.y * BM;
    const int n0   = blockIdx.x * BN;
    const int half = lane >> 4;          // 0 or 1
    const int lrow = lane & 15;

    v8f acc[4][4];
#pragma unroll
    for (int i = 0; i < 4; ++i)
#pragma unroll
        for (int j = 0; j < 4; ++j) {
            v8f z = {0.f, 0.f, 0.f, 0.f, 0.f, 0.f, 0.f, 0.f};
            acc[i][j] = z;
        }

    for (int kt = 0; kt < 1024; kt += BKK) {
        // ---- stage A tile (128 rows m x 32 cols k) as bf16 ----
        if (AMODE == 0) {
            const float* A = (const float*)Av;
            const int r  = tid >> 1;             // 0..127
            const int c0 = (tid & 1) * 16;       // 0 / 16
            const float4* src = (const float4*)(A + (size_t)(m0 + r) * 1024 + kt + c0);
            unsigned int* dst = (unsigned int*)&As[r * LDR + c0];
            if (kt + BKK < 1024)
                __builtin_prefetch((const char*)(src) + BKK * 4, 0, 1);
#pragma unroll
            for (int q = 0; q < 4; ++q) {
                float4 f = src[q];
                dst[2 * q]     = pack2bf(f.x, f.y);
                dst[2 * q + 1] = pack2bf(f.z, f.w);
            }
        } else {
            // bf16 source, contiguous along l; pack (k,k+1) pairs -> b32 store
            const unsigned short* A = (const unsigned short*)Av;
            const int kp = tid >> 4;             // 0..15 -> k = 2*kp
            const int l0 = (tid & 15) * 8;       // 0..120
            const unsigned short* src0 =
                A + ((size_t)((m0 >> 11) * 1024 + kt + 2 * kp)) * 2048 + (m0 & 2047) + l0;
            const unsigned short* src1 = src0 + 2048;
            if (kt + BKK < 1024)
                __builtin_prefetch((const char*)(src0) + (size_t)BKK * 2048 * 2, 0, 1);
            U4U u0; u0.v = *(const uint4*)src0;
            U4U u1; u1.v = *(const uint4*)src1;
#pragma unroll
            for (int i = 0; i < 8; ++i)
                *(unsigned int*)&As[(l0 + i) * LDR + 2 * kp] =
                    (unsigned int)u0.us[i] | ((unsigned int)u1.us[i] << 16);
        }
        // ---- stage W tile (32 k x 256 n) transposed, k-pairs packed ----
        {
            const int r2 = tid >> 4;             // 0..15 (k-pair index)
            const int c0 = (tid & 15) * 16;      // 16 columns per thread
            const float* w0 = W + (size_t)(kt + 2 * r2) * 1024 + n0 + c0;
            const float* w1 = w0 + 1024;
            if (kt + BKK < 1024)
                __builtin_prefetch((const char*)(w0) + (size_t)BKK * 1024 * 4, 0, 1);
#pragma unroll
            for (int h = 0; h < 4; ++h) {
                float4 a = ((const float4*)w0)[h];
                float4 b = ((const float4*)w1)[h];
                const int c = c0 + h * 4;
                *(unsigned int*)&BsT[(c + 0) * LDR + 2 * r2] = pack2bf(a.x, b.x);
                *(unsigned int*)&BsT[(c + 1) * LDR + 2 * r2] = pack2bf(a.y, b.y);
                *(unsigned int*)&BsT[(c + 2) * LDR + 2 * r2] = pack2bf(a.z, b.z);
                *(unsigned int*)&BsT[(c + 3) * LDR + 2 * r2] = pack2bf(a.w, b.w);
            }
        }
        __syncthreads();

        // ---- fragments: all contiguous 16B chunks -> ds_load_b128 ----
        BFrag afr[4], bfr[4];
#pragma unroll
        for (int tm = 0; tm < 4; ++tm) {
            const unsigned short* ap = &As[(wm * 64 + tm * 16 + lrow) * LDR];
            afr[tm].q[0] = *(const uint4*)(ap + half * 8);        // k = h8..h8+7
            afr[tm].q[1] = *(const uint4*)(ap + 16 + half * 8);   // k = 16+h8..
        }
#pragma unroll
        for (int tn = 0; tn < 4; ++tn) {
            const unsigned short* bp =
                &BsT[(wn * 64 + tn * 16 + lrow) * LDR + half * 8];
            bfr[tn].q[0] = *(const uint4*)(bp);                   // k = h8..h8+7
            bfr[tn].q[1] = *(const uint4*)(bp + 16);              // k = 16+h8..
        }
#pragma unroll
        for (int tm = 0; tm < 4; ++tm)
#pragma unroll
            for (int tn = 0; tn < 4; ++tn)
                acc[tm][tn] = __builtin_amdgcn_wmma_f32_16x16x32_bf16(
                    false, afr[tm].bf, false, bfr[tn].bf,
                    (short)0, acc[tm][tn], false, false);
        __syncthreads();
    }

    // ---- epilogue: bias + store (C layout: VGPR v -> M = v + 8*half) ----
#pragma unroll
    for (int tm = 0; tm < 4; ++tm) {
#pragma unroll
        for (int tn = 0; tn < 4; ++tn) {
            const int n = n0 + wn * 64 + tn * 16 + lrow;
            const float bvv = bias[n];
#pragma unroll
            for (int v = 0; v < 8; ++v) {
                const int m = m0 + wm * 64 + tm * 16 + v + (half << 3);
                const float val = acc[tm][tn][v] + bvv;
                size_t idx;
                if (OMODE == 0) idx = (size_t)m * 1024 + n;
                else {
                    const int b = m >> 11, l = m & 2047;
                    idx = ((size_t)(b * 1024 + n)) * 2048 + l;
                }
                Out[idx] = val;
            }
        }
    }
}

// --------------------------- FFT autocorrelation ---------------------------
__device__ void fft2048(float2* z, float sign, int tid)
{
    for (int i = tid; i < 2048; i += 256) {                 // bit-reverse (11b)
        const int j = (int)(__brev((unsigned)i) >> 21);
        if (j > i) { float2 t = z[i]; z[i] = z[j]; z[j] = t; }
    }
    __syncthreads();
    int lg = 0;
    for (int len = 2; len <= 2048; len <<= 1, ++lg) {
        const int hmask = (len >> 1) - 1;
        const float ang0 = sign * 6.2831853071795864f / (float)len;
        for (int b = tid; b < 1024; b += 256) {
            const int grp = b >> lg;
            const int pos = b & hmask;
            const int i = (grp << (lg + 1)) + pos;
            const int j = i + (hmask + 1);
            float s, c;
            __sincosf(ang0 * (float)pos, &s, &c);
            const float2 u = z[i], v = z[j];
            const float2 t = { v.x * c - v.y * s, v.x * s + v.y * c };
            z[i] = make_float2(u.x + t.x, u.y + t.y);
            z[j] = make_float2(u.x - t.x, u.y - t.y);
        }
        __syncthreads();
    }
}

__global__ __launch_bounds__(256)
void autocorr_fft_topk(const float* qb, const float* kb, const float* vb,
                       unsigned short* aggb /* bf16 out, aliases qb region */)
{
    __shared__ float2 z[2048];
    __shared__ float  vloc[2048];
    __shared__ float  corrv[2048];
    __shared__ float  redv[256];
    __shared__ int    redi[256];
    __shared__ float  topv[7];
    __shared__ int    topi[7];
    __shared__ float  wgt[7];

    const int tid = threadIdx.x;
    const size_t base = (size_t)blockIdx.x * 2048;

    for (int i = tid; i < 2048; i += 256) {
        z[i] = make_float2(qb[base + i], kb[base + i]);     // pack q + i*k
        vloc[i] = vb[base + i];
    }
    __syncthreads();

    fft2048(z, -1.0f, tid);                                 // forward FFT

    for (int f = tid; f <= 1024; f += 256) {                // Hermitian split
        const float2 Zf = z[f];
        const float2 Zm = z[(2048 - f) & 2047];
        const float qr = 0.5f * (Zf.x + Zm.x);
        const float qi = 0.5f * (Zf.y - Zm.y);
        const float kr = 0.5f * (Zf.y + Zm.y);
        const float ki = -0.5f * (Zf.x - Zm.x);
        const float pr = qr * kr + qi * ki;                 // P = Q * conj(K)
        const float pi = qi * kr - qr * ki;
        z[f] = make_float2(pr, pi);
        if (f != 0 && f != 1024) z[2048 - f] = make_float2(pr, -pi);
    }
    __syncthreads();

    fft2048(z, +1.0f, tid);                                 // inverse FFT
    for (int i = tid; i < 2048; i += 256)
        corrv[i] = z[i].x * (1.0f / 2048.0f);
    __syncthreads();

    for (int it = 0; it < 7; ++it) {                        // top-7 argmax
        float best = -1e30f; int bi = 0;
        for (int i = tid; i < 2048; i += 256) {
            const float v = corrv[i];
            if (v > best) { best = v; bi = i; }
        }
        redv[tid] = best; redi[tid] = bi;
        __syncthreads();
        for (int off = 128; off > 0; off >>= 1) {
            if (tid < off && redv[tid + off] > redv[tid]) {
                redv[tid] = redv[tid + off]; redi[tid] = redi[tid + off];
            }
            __syncthreads();
        }
        if (tid == 0) {
            topv[it] = redv[0]; topi[it] = redi[0];
            corrv[redi[0]] = -1e30f;
        }
        __syncthreads();
    }

    if (tid == 0) {                                         // softmax over 7
        const float mx = topv[0];
        float s = 0.f, e[7];
#pragma unroll
        for (int i = 0; i < 7; ++i) { e[i] = __expf(topv[i] - mx); s += e[i]; }
        const float inv = 1.0f / s;
#pragma unroll
        for (int i = 0; i < 7; ++i) wgt[i] = e[i] * inv;
    }
    __syncthreads();

    for (int l = tid; l < 2048; l += 256) {                 // roll-gather
        float a = 0.f;
#pragma unroll
        for (int i = 0; i < 7; ++i)
            a += wgt[i] * vloc[(l + topi[i]) & 2047];
        aggb[base + l] = f2bf(a);                           // bf16 agg
    }
}

// ------------------------------- launcher ----------------------------------
extern "C" void kernel_launch(void* const* d_in, const int* in_sizes, int n_in,
                              void* d_out, int out_size, void* d_ws, size_t ws_size,
                              hipStream_t stream)
{
    const float* queries = (const float*)d_in[0];
    const float* keys    = (const float*)d_in[1];
    const float* values  = (const float*)d_in[2];
    const float* Wq = (const float*)d_in[3];
    const float* bq = (const float*)d_in[4];
    const float* Wk = (const float*)d_in[5];
    const float* bk = (const float*)d_in[6];
    const float* Wv = (const float*)d_in[7];
    const float* bv = (const float*)d_in[8];
    const float* Wo = (const float*)d_in[9];
    const float* bo = (const float*)d_in[10];
    float* out = (float*)d_out;

    const size_t SER = (size_t)16 * 1024 * 2048;   // B*D*L floats
    float* qbuf = (float*)d_ws;
    float* kbuf = qbuf + SER;
    float* vbuf = kbuf + SER;
    unsigned short* aggbf = (unsigned short*)qbuf; // bf16 agg over qbuf region

    dim3 grid(1024 / BN, 32768 / BM);              // 4 x 256 tiles

    wmma_gemm_bias<0, 1><<<grid, 256, 0, stream>>>(queries, Wq, bq, qbuf);
    wmma_gemm_bias<0, 1><<<grid, 256, 0, stream>>>(keys,    Wk, bk, kbuf);
    wmma_gemm_bias<0, 1><<<grid, 256, 0, stream>>>(values,  Wv, bv, vbuf);

    autocorr_fft_topk<<<16 * 16 * 64, 256, 0, stream>>>(qbuf, kbuf, vbuf, aggbf);

    wmma_gemm_bias<1, 0><<<grid, 256, 0, stream>>>(aggbf, Wo, bo, out);
}
